// BabyBrain_87952340288080
// MI455X (gfx1250) — matrix operand order repaired and checked by reading, not compiled, not run on hardware
//
#include <hip/hip_runtime.h>

// ---------------- model dims (match reference) ----------------
#define B_SZ     2
#define L_SEQ    512
#define BL       (B_SZ * L_SEQ)        // 1024 token rows
#define VOCAB    512
#define D_MODEL  768
#define N_LAYERS 12
#define D_STATE  16
#define D_CONV   4
#define D_INNER  (2 * D_MODEL)         // 1536
#define DT_RANK  48
#define XDBL_W   (DT_RANK + 2 * D_STATE)  // 80

// ---------------- WMMA vector types (gfx1250, wave32) ----------------
typedef __attribute__((ext_vector_type(16))) __bf16    v16bf;
typedef __attribute__((ext_vector_type(8)))  float     v8f;
typedef __attribute__((ext_vector_type(8)))  unsigned  v8u;

static __device__ __forceinline__ int imin(int a, int b) { return a < b ? a : b; }

// pack two f32 into a packed bf16 pair (round-half-up)
static __device__ __forceinline__ unsigned pk2(float a, float b) {
  union { float f; unsigned u; } x, y;
  x.f = a; y.f = b;
  return __byte_perm(x.u + 0x8000u, y.u + 0x8000u, 0x7632);
}

static __device__ __forceinline__ __bf16 f2b(float f) {
  union { float f; unsigned u; } a; a.f = f;
  union { unsigned short s; __bf16 b; } o;
  o.s = (unsigned short)((a.u + 0x8000u) >> 16);
  return o.b;
}

static __device__ __forceinline__ float siluf(float x) {
  return x / (1.0f + __expf(-x));
}

static __device__ __forceinline__ float softplusf(float x) {
  return (x > 20.0f) ? x : log1pf(__expf(x));
}

#define WMMA_BF16(Af, Bf, Acc) \
  __builtin_amdgcn_wmma_f32_16x16x32_bf16(false, (Af), false, (Bf), (short)0, (Acc), false, false)

// A fragment (16x32 bf16, ISA 7.12.2): two unguarded b128 loads, no VALU.
static __device__ __forceinline__ v16bf load_a_bf(const __bf16* __restrict__ p) {
  uint4 lo = *(const uint4*)(p);        // K +0..7
  uint4 hi = *(const uint4*)(p + 16);   // K +16..23
  v8u au;
  au[0] = lo.x; au[1] = lo.y; au[2] = lo.z; au[3] = lo.w;
  au[4] = hi.x; au[5] = hi.y; au[6] = hi.z; au[7] = hi.w;
  return __builtin_bit_cast(v16bf, au);
}

// K-tail A fragment: low 8 K valid, K+16..31 zero
static __device__ __forceinline__ v16bf load_a_bf_tail(const __bf16* __restrict__ p) {
  uint4 lo = *(const uint4*)(p);
  v8u au;
  au[0] = lo.x; au[1] = lo.y; au[2] = lo.z; au[3] = lo.w;
  au[4] = 0u;   au[5] = 0u;   au[6] = 0u;   au[7] = 0u;
  return __builtin_bit_cast(v16bf, au);
}

static __device__ __forceinline__ void store_tile(
    float* __restrict__ C, const float* __restrict__ bias,
    const float* __restrict__ addend, int act, int ldc,
    int row, int col, v8f acc)
{
  const float bv = bias ? bias[col] : 0.0f;
#pragma unroll
  for (int j = 0; j < 8; ++j) {
    const size_t idx = (size_t)(row + j) * ldc + col;
    float v = acc[j] + bv;
    if (addend) v += addend[idx];
    if (act == 1) v = softplusf(v);
    C[idx] = v;
  }
}

// =====================================================================
// f32 -> packed bf16 converter: each thread converts 8 values
// (2 x global_load_b128 -> 4 x v_perm-class packs -> 1 x b128 store).
// n8 = element_count / 8 (all operand sizes here are multiples of 8).
// =====================================================================
__global__ __launch_bounds__(256)
void mamba_cvt_bf16(const float* __restrict__ in, unsigned* __restrict__ outp, int n8)
{
  const int i = blockIdx.x * 256 + threadIdx.x;
  if (i < n8) {
    const float4 f0 = ((const float4*)in)[2 * i];
    const float4 f1 = ((const float4*)in)[2 * i + 1];
    uint4 o;
    o.x = pk2(f0.x, f0.y); o.y = pk2(f0.z, f0.w);
    o.z = pk2(f1.x, f1.y); o.w = pk2(f1.z, f1.w);
    ((uint4*)outp)[i] = o;
  }
}

// =====================================================================
// Fused GEMM:  C[M,N] = act( A[M,K] @ W[N,K]^T + bias + addend )
// All operands pre-converted bf16; WMMA 16x16x32 bf16, f32 accumulate.
//   - 256 threads = 8 waves; each wave owns 32(M) x 64(N)
//     (8 accumulators) -> block tile 256(M) x 64(N).
//   - B tile (64 cols x 32 K bf16, 4 KB) double-buffered in LDS:
//     next slab staged (b128 copy, no conversion) while current slab
//     feeds 8 WMMAs; ONE barrier per K-step.
//   - Hot loop has zero conversion VALU: 4 b128 A loads, 1 b128 B copy,
//     8 ds_load_b128, 8 v_wmma.
// Requirements (true at every call site): M % 256 == 0, N % 16 == 0,
// K % 16 == 0, K >= 32. EXEC all-1s wherever WMMA executes.
// act: 0 = identity, 1 = softplus.
// =====================================================================
__global__ __launch_bounds__(256)
void mamba_wmma_gemm(const __bf16* __restrict__ A, int lda,
                     const __bf16* __restrict__ W, int ldw,
                     float* __restrict__ C, int ldc,
                     const float* __restrict__ bias,
                     const float* __restrict__ addend,
                     int M, int N, int K, int act)
{
  __shared__ unsigned sB[2][64 * 16];   // 2 x 4 KB double buffer

  const int tid  = threadIdx.x;
  const int wave = tid >> 5;                    // 0..7  (M direction)
  const int lane = tid & 31;                    // wave32
  const int row0 = blockIdx.y * 256 + wave * 32;
  const int col0 = blockIdx.x * 64;

  const int am  = lane & 15;
  const int akb = (lane >> 4) << 3;             // 0 or 8
  const int bn  = lane & 15;
  const int bks = (lane >> 4) << 4;             // 0 or 16

  // staging task: each thread copies 8 bf16 (16B) of one column per step
  const int scol = tid >> 2;                    // 0..63
  const int skc  = (tid & 3) << 3;              // 0,8,16,24

  const __bf16* __restrict__ Arow0 = A + (size_t)(row0 + am) * lda;
  const __bf16* __restrict__ Arow1 = Arow0 + (size_t)16 * lda;
  const __bf16* __restrict__ Wst   = W + (size_t)imin(col0 + scol, N - 1) * ldw;
  unsigned* __restrict__ sSt = &sB[0][0] + scol * 16 + (((tid & 1) << 3) | ((tid & 2) << 1));

  const unsigned koff = (unsigned)(bks >> 1);
  const unsigned* __restrict__ sBr = &sB[0][0] + bn * 16 + koff;

  v8f z = {0,0,0,0,0,0,0,0};
  v8f a00 = z, a01 = z, a02 = z, a03 = z;       // row tile 0
  v8f a10 = z, a11 = z, a12 = z, a13 = z;       // row tile 1

  const int K32 = K & ~31;

  // ---- prologue: stage first slab into buffer 0 ----
  *(uint4*)sSt = *(const uint4*)(Wst + skc);
  __syncthreads();

  for (int k0 = 0; k0 < K32; k0 += 32) {
    const int cur = (k0 >> 5) & 1;
    const int kn  = k0 + 32;

    // ---- stage next slab into the other buffer (overlaps WMMAs) ----
    if (kn < K32) {
      __builtin_prefetch(Wst + kn + 32 + skc, 0, 1);   // global_prefetch_b8
      *(uint4*)(sSt + (cur ^ 1) * 1024) = *(const uint4*)(Wst + kn + skc);
    } else if (kn < K) {                               // 16-wide K tail slab
      uint4 t = {0u, 0u, 0u, 0u};
      if (kn + skc < K) t = *(const uint4*)(Wst + kn + skc);  // chunk-uniform
      *(uint4*)(sSt + (cur ^ 1) * 1024) = t;
    }

    // ---- operand fragments (all loads issued before any WMMA) ----
    const unsigned* src = sBr + cur * 1024;
    const v16bf af0 = load_a_bf(Arow0 + k0 + akb);
    const v16bf af1 = load_a_bf(Arow1 + k0 + akb);
    const v16bf bf0 = *(const v16bf*)(src);
    const v16bf bf1 = *(const v16bf*)(src + 256);
    const v16bf bf2 = *(const v16bf*)(src + 512);
    const v16bf bf3 = *(const v16bf*)(src + 768);

    a00 = WMMA_BF16(af0, bf0, a00);  a10 = WMMA_BF16(af1, bf0, a10);
    a01 = WMMA_BF16(af0, bf1, a01);  a11 = WMMA_BF16(af1, bf1, a11);
    a02 = WMMA_BF16(af0, bf2, a02);  a12 = WMMA_BF16(af1, bf2, a12);
    a03 = WMMA_BF16(af0, bf3, a03);  a13 = WMMA_BF16(af1, bf3, a13);
    __syncthreads();
  }

  // ---- 16-wide K tail (K % 32 == 16, e.g. dt_proj K=48) ----
  if (K32 < K) {
    const unsigned* src = sBr + ((K32 >> 5) & 1) * 1024;  // staged in last iter
    const v16bf af0 = load_a_bf_tail(Arow0 + K32 + akb);
    const v16bf af1 = load_a_bf_tail(Arow1 + K32 + akb);
    const v16bf bf0 = *(const v16bf*)(src);
    const v16bf bf1 = *(const v16bf*)(src + 256);
    const v16bf bf2 = *(const v16bf*)(src + 512);
    const v16bf bf3 = *(const v16bf*)(src + 768);

    a00 = WMMA_BF16(af0, bf0, a00);  a10 = WMMA_BF16(af1, bf0, a10);
    a01 = WMMA_BF16(af0, bf1, a01);  a11 = WMMA_BF16(af1, bf1, a11);
    a02 = WMMA_BF16(af0, bf2, a02);  a12 = WMMA_BF16(af1, bf2, a12);
    a03 = WMMA_BF16(af0, bf3, a03);  a13 = WMMA_BF16(af1, bf3, a13);
  }

  // ---- store: C layout (ISA 7.12.2): lanes 0-15 -> M=vgpr, N=lane;
  //      lanes 16-31 -> M=8+vgpr, N=lane-16. Guards are wave-uniform. ----
  const int crow0 = row0 + ((lane >> 4) << 3);
  const int cn    = lane & 15;
  if (col0 < N) {
    store_tile(C, bias, addend, act, ldc, crow0,      col0 + cn,      a00);
    store_tile(C, bias, addend, act, ldc, crow0 + 16, col0 + cn,      a10);
  }
  if (col0 + 16 < N) {
    store_tile(C, bias, addend, act, ldc, crow0,      col0 + 16 + cn, a01);
    store_tile(C, bias, addend, act, ldc, crow0 + 16, col0 + 16 + cn, a11);
  }
  if (col0 + 32 < N) {
    store_tile(C, bias, addend, act, ldc, crow0,      col0 + 32 + cn, a02);
    store_tile(C, bias, addend, act, ldc, crow0 + 16, col0 + 32 + cn, a12);
  }
  if (col0 + 48 < N) {
    store_tile(C, bias, addend, act, ldc, crow0,      col0 + 48 + cn, a03);
    store_tile(C, bias, addend, act, ldc, crow0 + 16, col0 + 48 + cn, a13);
  }
}

// =====================================================================
// Embedding gather: x[bl,:] = emb[ids[bl],:]
// =====================================================================
__global__ __launch_bounds__(256)
void mamba_embed(const float* __restrict__ emb, const int* __restrict__ ids,
                 float* __restrict__ x)
{
  const int bl  = blockIdx.x;
  const int tok = ids[bl];
  for (int j = threadIdx.x; j < D_MODEL; j += 256)
    x[(size_t)bl * D_MODEL + j] = emb[(size_t)tok * D_MODEL + j];
}

// =====================================================================
// Causal depthwise conv (D_CONV=4) + bias + SiLU.
// Writes f32 u (for the scan) and its bf16 mirror (for x_proj GEMM).
// =====================================================================
__global__ __launch_bounds__(256)
void mamba_conv_silu(const float* __restrict__ xr,
                     const float* __restrict__ cw,   // [D_INNER,4]
                     const float* __restrict__ cb,   // [D_INNER]
                     float* __restrict__ u,          // [BL, D_INNER]
                     __bf16* __restrict__ ubf)       // bf16 mirror
{
  const int bl = blockIdx.x;
  const int b  = bl / L_SEQ;
  const int l  = bl % L_SEQ;
  for (int j = 0; j < D_INNER; j += 256) {
    const int d = j + threadIdx.x;
    const float4 w = *(const float4*)(cw + d * D_CONV);
    float acc = cb[d];
    const float* col = xr + (size_t)b * L_SEQ * (2 * D_INNER) + d;
    if (l >= 3) {
      acc += w.x * col[(size_t)(l - 3) * (2 * D_INNER)];
      acc += w.y * col[(size_t)(l - 2) * (2 * D_INNER)];
      acc += w.z * col[(size_t)(l - 1) * (2 * D_INNER)];
    } else {
      if (l >= 1) acc += w.z * col[(size_t)(l - 1) * (2 * D_INNER)];
      if (l >= 2) acc += w.y * col[(size_t)(l - 2) * (2 * D_INNER)];
    }
    acc += w.w * col[(size_t)l * (2 * D_INNER)];
    const float v = siluf(acc);
    u  [(size_t)bl * D_INNER + d] = v;
    ubf[(size_t)bl * D_INNER + d] = f2b(v);
  }
}

// =====================================================================
// Selective scan. One lane per (b,d) channel; 16-wide SSM state in VGPRs.
// Bm/Cm (shared per timestep) broadcast through LDS. 64 threads/block.
// =====================================================================
__global__ __launch_bounds__(64)
void mamba_scan(const float* __restrict__ u,      // [B,L,DI]
                const float* __restrict__ delta,  // [B,L,DI]
                const float* __restrict__ xdbl,   // [B,L,80]  (Bm@48, Cm@64)
                const float* __restrict__ A_log,  // [DI,16]
                const float* __restrict__ Dp,     // [DI]
                float* __restrict__ y)            // [B,L,DI]
{
  const int b = blockIdx.y;
  const int d = blockIdx.x * 64 + threadIdx.x;

  __shared__ float sB[D_STATE];
  __shared__ float sC[D_STATE];

  float h[D_STATE], Arow[D_STATE];
#pragma unroll
  for (int n = 0; n < D_STATE; ++n) {
    h[n]    = 0.0f;
    Arow[n] = -__expf(A_log[(size_t)d * D_STATE + n]);
  }
  const float Dv    = Dp[d];
  const size_t base = (size_t)b * L_SEQ;

  for (int l = 0; l < L_SEQ; ++l) {
    if (threadIdx.x < 32) {
      const float* src = xdbl + (base + l) * XDBL_W + DT_RANK;
      if (threadIdx.x < 16) sB[threadIdx.x]      = src[threadIdx.x];
      else                  sC[threadIdx.x - 16] = src[threadIdx.x];
    }
    __syncthreads();
    const float dlt = delta[(base + l) * D_INNER + d];
    const float uu  = u    [(base + l) * D_INNER + d];
    float acc = 0.0f;
#pragma unroll
    for (int n = 0; n < D_STATE; ++n) {
      const float dA = __expf(dlt * Arow[n]);
      h[n] = dA * h[n] + dlt * sB[n] * uu;
      acc += h[n] * sC[n];
    }
    y[(base + l) * D_INNER + d] = acc + uu * Dv;
    __syncthreads();   // protect sB/sC before next timestep's refill
  }
}

// =====================================================================
// Gating: yg_bf = bf16( y * silu(res) ) — consumed only by out_proj GEMM.
// =====================================================================
__global__ __launch_bounds__(256)
void mamba_gate(const float* __restrict__ y, const float* __restrict__ xr,
                __bf16* __restrict__ yg, int total)
{
  const int i = blockIdx.x * 256 + threadIdx.x;
  if (i < total) {
    const int d  = i % D_INNER;
    const int bl = i / D_INNER;
    const float r = xr[(size_t)bl * (2 * D_INNER) + D_INNER + d];
    yg[i] = f2b(y[i] * siluf(r));
  }
}

// =====================================================================
// LayerNorm over D_MODEL; writes f32 hidden (model output) + bf16 mirror
// (head GEMM input).
// =====================================================================
__global__ __launch_bounds__(256)
void mamba_layernorm(const float* __restrict__ x, const float* __restrict__ g,
                     const float* __restrict__ bta,
                     float* __restrict__ hidden, __bf16* __restrict__ hbf)
{
  __shared__ float red[256];
  const int bl = blockIdx.x;
  const float* row = x + (size_t)bl * D_MODEL;

  float s = 0.0f;
  for (int j = threadIdx.x; j < D_MODEL; j += 256) s += row[j];
  red[threadIdx.x] = s; __syncthreads();
  for (int off = 128; off > 0; off >>= 1) {
    if (threadIdx.x < off) red[threadIdx.x] += red[threadIdx.x + off];
    __syncthreads();
  }
  const float mu = red[0] / (float)D_MODEL;
  __syncthreads();

  float v = 0.0f;
  for (int j = threadIdx.x; j < D_MODEL; j += 256) {
    const float t = row[j] - mu; v += t * t;
  }
  red[threadIdx.x] = v; __syncthreads();
  for (int off = 128; off > 0; off >>= 1) {
    if (threadIdx.x < off) red[threadIdx.x] += red[threadIdx.x + off];
    __syncthreads();
  }
  const float rstd = rsqrtf(red[0] / (float)D_MODEL + 1e-5f);

  for (int j = threadIdx.x; j < D_MODEL; j += 256) {
    const float o = (row[j] - mu) * rstd * g[j] + bta[j];
    hidden[(size_t)bl * D_MODEL + j] = o;
    hbf   [(size_t)bl * D_MODEL + j] = f2b(o);
  }
}

// =====================================================================
// Host-side orchestration
// =====================================================================
static inline void cvt(const float* in, __bf16* out, int n, hipStream_t s) {
  const int n8 = n / 8;                 // all operand sizes are multiples of 8
  mamba_cvt_bf16<<<(n8 + 255) / 256, 256, 0, s>>>(in, (unsigned*)out, n8);
}

extern "C" void kernel_launch(void* const* d_in, const int* in_sizes, int n_in,
                              void* d_out, int out_size, void* d_ws, size_t ws_size,
                              hipStream_t stream)
{
  (void)in_sizes; (void)n_in; (void)out_size; (void)ws_size;

  const float* emb    = (const float*)d_in[0];
  const float* in_w   = (const float*)d_in[1];
  const float* conv_w = (const float*)d_in[2];
  const float* conv_b = (const float*)d_in[3];
  const float* xp_w   = (const float*)d_in[4];
  const float* dt_w   = (const float*)d_in[5];
  const float* dt_b   = (const float*)d_in[6];
  const float* A_log  = (const float*)d_in[7];
  const float* Dp     = (const float*)d_in[8];
  const float* out_w  = (const float*)d_in[9];
  const float* ln_g   = (const float*)d_in[10];
  const float* ln_b   = (const float*)d_in[11];
  const float* head_w = (const float*)d_in[12];
  const float* head_b = (const float*)d_in[13];
  const int*   ids    = (const int*)d_in[14];

  // ---- workspace carve-up: ~35 MB f32 + ~14 MB bf16 ----
  float* ws   = (float*)d_ws;
  float* x    = ws;                                  // [BL, D_MODEL]
  float* xr   = x    + (size_t)BL * D_MODEL;         // [BL, 2*D_INNER]
  float* u    = xr   + (size_t)BL * 2 * D_INNER;     // [BL, D_INNER]
  float* xdbl = u    + (size_t)BL * D_INNER;         // [BL, 80]
  float* dlt  = xdbl + (size_t)BL * XDBL_W;          // [BL, D_INNER]
  float* ybuf = dlt  + (size_t)BL * D_INNER;         // [BL, D_INNER]

  __bf16* x_bf    = (__bf16*)(ybuf + (size_t)BL * D_INNER);
  __bf16* u_bf    = x_bf    + (size_t)BL * D_MODEL;
  __bf16* xdbl_bf = u_bf    + (size_t)BL * D_INNER;
  __bf16* yg_bf   = xdbl_bf + (size_t)BL * XDBL_W;
  __bf16* hid_bf  = yg_bf   + (size_t)BL * D_INNER;
  __bf16* w_bf    = hid_bf  + (size_t)BL * D_MODEL;  // max 2*D_INNER*D_MODEL elems

  float* logits = (float*)d_out;                     // [BL, VOCAB]
  float* hidden = logits + (size_t)BL * VOCAB;       // [BL, D_MODEL]

  const dim3 blk(256);

  mamba_embed<<<BL, blk, 0, stream>>>(emb, ids, x);

  for (int i = 0; i < N_LAYERS; ++i) {
    // residual stream -> bf16 (fresh each layer; x updated by out_proj)
    cvt(x, x_bf, BL * D_MODEL, stream);

    // in_proj: xr = x @ in_w^T   [1024 x 3072], K=768
    cvt(in_w + (size_t)i * 2 * D_INNER * D_MODEL, w_bf, 2 * D_INNER * D_MODEL, stream);
    mamba_wmma_gemm<<<dim3(48, 4), blk, 0, stream>>>(
        x_bf, D_MODEL, w_bf, D_MODEL,
        xr, 2 * D_INNER, nullptr, nullptr, BL, 2 * D_INNER, D_MODEL, 0);

    mamba_conv_silu<<<BL, blk, 0, stream>>>(
        xr, conv_w + (size_t)i * D_INNER * D_CONV, conv_b + (size_t)i * D_INNER,
        u, u_bf);

    // x_proj: xdbl = u @ xp_w^T   [1024 x 80], K=1536
    cvt(xp_w + (size_t)i * XDBL_W * D_INNER, w_bf, XDBL_W * D_INNER, stream);
    mamba_wmma_gemm<<<dim3(2, 4), blk, 0, stream>>>(
        u_bf, D_INNER, w_bf, D_INNER,
        xdbl, XDBL_W, nullptr, nullptr, BL, XDBL_W, D_INNER, 0);
    cvt(xdbl, xdbl_bf, BL * XDBL_W, stream);

    // dt_proj + softplus: dlt = softplus(xdbl[:, :48] @ dt_w^T + dt_b)  K=48
    cvt(dt_w + (size_t)i * D_INNER * DT_RANK, w_bf, D_INNER * DT_RANK, stream);
    mamba_wmma_gemm<<<dim3(24, 4), blk, 0, stream>>>(
        xdbl_bf, XDBL_W, w_bf, DT_RANK,
        dlt, D_INNER, dt_b + (size_t)i * D_INNER, nullptr, BL, D_INNER, DT_RANK, 1);

    mamba_scan<<<dim3(D_INNER / 64, B_SZ), dim3(64), 0, stream>>>(
        u, dlt, xdbl, A_log + (size_t)i * D_INNER * D_STATE,
        Dp + (size_t)i * D_INNER, ybuf);

    mamba_gate<<<(BL * D_INNER) / 256, blk, 0, stream>>>(
        ybuf, xr, yg_bf, BL * D_INNER);

    // out_proj + residual: x = yg @ out_w^T + x   [1024 x 768], K=1536
    cvt(out_w + (size_t)i * D_MODEL * D_INNER, w_bf, D_MODEL * D_INNER, stream);
    mamba_wmma_gemm<<<dim3(12, 4), blk, 0, stream>>>(
        yg_bf, D_INNER, w_bf, D_INNER,
        x, D_MODEL, nullptr, x, BL, D_MODEL, D_INNER, 0);
  }

  mamba_layernorm<<<BL, blk, 0, stream>>>(x, ln_g, ln_b, hidden, hid_bf);

  // head: logits = hidden @ head_w^T + head_b   [1024 x 512], K=768
  cvt(head_w, w_bf, VOCAB * D_MODEL, stream);
  mamba_wmma_gemm<<<dim3(8, 4), blk, 0, stream>>>(
      hid_bf, D_MODEL, w_bf, D_MODEL,
      logits, VOCAB, head_b, nullptr, BL, VOCAB, D_MODEL, 0);
}
